// GraphConvPosEnc_7043746365719
// MI455X (gfx1250) — matrix operand order repaired and checked by reading, not compiled, not run on hardware
//
#include <hip/hip_runtime.h>
#include <hip/hip_bf16.h>
#include <hip/hip_fp16.h>

typedef __attribute__((ext_vector_type(2)))  float    v2f;
typedef __attribute__((ext_vector_type(8)))  float    v8f;
typedef __attribute__((ext_vector_type(16))) _Float16 v16h;

#define DFEAT 128   // D_OUT and D_IN_X
#define DIN   256   // concat dim

// ---------------------------------------------------------------------------
// Kernel 1: x_proj = concat(x,state) @ W_in + b_in  (fp32 WMMA 16x16x4)
// One wave computes one 16x16 tile of x_proj. 8 waves/block = 16 rows x 128 cols.
// A layout (f32 16x4): lane m=lane&15 holds row m, K = kb + 2*(lane>>4) + {0,1}
// B layout (f32 4x16): lane holds col n=lane&15, same K pair -> strided loads.
// K loop is split at the concat boundary (K=128) so the hot loop has no
// per-iteration source select.
// ---------------------------------------------------------------------------
__global__ __launch_bounds__(256) void proj_wmma_kernel(
    const float* __restrict__ x, const float* __restrict__ st,
    const float* __restrict__ W_in, const float* __restrict__ b_in,
    float* __restrict__ xp, int nrows)
{
  const int lane = threadIdx.x & 31;
  const int wave = threadIdx.x >> 5;
  const int rt   = blockIdx.x;        // row tile
  const int ct   = wave;              // col tile (8 tiles of 16 = 128)
  const int m    = lane & 15;
  const int h    = lane >> 4;
  const int col  = ct * 16 + m;

  const int row  = rt * 16 + m;
  const int rowc = row < nrows ? row : nrows - 1;
  const float* rx = x  + (size_t)rowc * DFEAT + 2 * h;   // this lane's K pair base
  const float* rs = st + (size_t)rowc * DFEAT + 2 * h;
  const float* wb = W_in + (size_t)(2 * h) * DFEAT + col;

  v8f acc;
  {
    const float bv = b_in[col];
#pragma unroll
    for (int r = 0; r < 8; ++r) acc[r] = bv;
  }

  // K = 0..127 from x
#pragma unroll 8
  for (int t = 0; t < 32; ++t) {
    v2f a, b;
    a.x = rx[t * 4];
    a.y = rx[t * 4 + 1];
    const float* bp = wb + (size_t)(t * 4) * DFEAT;
    b.x = bp[0];
    b.y = bp[DFEAT];
    acc = __builtin_amdgcn_wmma_f32_16x16x4_f32(false, a, false, b, (short)0, acc,
                                                false, false);
  }
  // K = 128..255 from state
#pragma unroll 8
  for (int t = 0; t < 32; ++t) {
    v2f a, b;
    a.x = rs[t * 4];
    a.y = rs[t * 4 + 1];
    const float* bp = wb + (size_t)(DFEAT + t * 4) * DFEAT;
    b.x = bp[0];
    b.y = bp[DFEAT];
    acc = __builtin_amdgcn_wmma_f32_16x16x4_f32(false, a, false, b, (short)0, acc,
                                                false, false);
  }

  // D layout: VGPR r -> row (rt*16 + r + 8h), col = n. Fast path: whole tile
  // in range (wave-uniform) -> unguarded stores from one base pointer.
  float* op = xp + (size_t)(rt * 16 + 8 * h) * DFEAT + col;
  if (rt * 16 + 16 <= nrows) {
#pragma unroll
    for (int r = 0; r < 8; ++r) op[(size_t)r * DFEAT] = acc[r];
  } else {
#pragma unroll
    for (int r = 0; r < 8; ++r)
      if (rt * 16 + 8 * h + r < nrows) op[(size_t)r * DFEAT] = acc[r];
  }
}

// ---------------------------------------------------------------------------
// Kernel 2: per-edge gather + MLP gate (f16 WMMA 16x16x32) + atomic scatter.
// One wave handles 16 edges. The 16x128 msg tile lives in registers in the
// f16 A-matrix layout: lane (m,h) holds, for chunk c, cols [32c+8h, +8) and
// [32c+16+8h, +8) of row m (contiguous -> float4 loads, L2-resident gather).
// ---------------------------------------------------------------------------
__global__ __launch_bounds__(256) void edge_wmma_kernel(
    const float* __restrict__ xp, const long long* __restrict__ ei,
    const float* __restrict__ ew, const float* __restrict__ W1,
    const float* __restrict__ b1, const float* __restrict__ W2,
    const float* __restrict__ b2, float* __restrict__ out,
    float* __restrict__ deg, int E, int ntiles)
{
  __shared__ float wlds[8][16];
  const int lane = threadIdx.x & 31;
  const int wave = threadIdx.x >> 5;
  const int tile = blockIdx.x * 8 + wave;
  if (tile >= ntiles) return;   // wave-uniform: EXEC stays all-1 for WMMA

  const int m  = lane & 15;
  const int h  = lane >> 4;
  const int e  = tile * 16 + m;
  const int ec = e < E ? e : E - 1;
  const int src = (int)ei[ec];
  const int dst = (int)ei[(size_t)E + ec];
  const float* mrow = xp + (size_t)src * DFEAT;

  float msg[64];
#pragma unroll
  for (int c = 0; c < 4; ++c) {
    const float4 f0 = *(const float4*)(mrow + 32 * c + 8 * h);
    const float4 f1 = *(const float4*)(mrow + 32 * c + 8 * h + 4);
    const float4 f2 = *(const float4*)(mrow + 32 * c + 16 + 8 * h);
    const float4 f3 = *(const float4*)(mrow + 32 * c + 16 + 8 * h + 4);
    msg[c*16 +  0] = f0.x; msg[c*16 +  1] = f0.y; msg[c*16 +  2] = f0.z; msg[c*16 +  3] = f0.w;
    msg[c*16 +  4] = f1.x; msg[c*16 +  5] = f1.y; msg[c*16 +  6] = f1.z; msg[c*16 +  7] = f1.w;
    msg[c*16 +  8] = f2.x; msg[c*16 +  9] = f2.y; msg[c*16 + 10] = f2.z; msg[c*16 + 11] = f2.w;
    msg[c*16 + 12] = f3.x; msg[c*16 + 13] = f3.y; msg[c*16 + 14] = f3.z; msg[c*16 + 15] = f3.w;
  }

  // h_edge = msg @ W1 + b1 : C tile = [16 edges x 16 hidden]
  v8f acc;
  {
    const float bv = b1[m];
#pragma unroll
    for (int r = 0; r < 8; ++r) acc[r] = bv;
  }
#pragma unroll
  for (int c = 0; c < 4; ++c) {
    v16h a;
#pragma unroll
    for (int j = 0; j < 16; ++j) a[j] = (_Float16)msg[c * 16 + j];
    // B layout (f16 32x16): lane half h holds K=8h..8h+7 and 16+8h..16+8h+7 (matches A)
    v16h b;
    const float* wp = W1 + (size_t)(32 * c + 8 * h) * 16 + m;
#pragma unroll
    for (int j = 0; j < 16; ++j) b[j] = (_Float16)wp[(j < 8 ? j : j + 8) * 16];
    acc = __builtin_amdgcn_wmma_f32_16x16x32_f16(false, a, false, b, (short)0, acc,
                                                 false, false);
  }

  // gate: leaky_relu -> dot W2 (xor-reduce over 16-lane halves) -> sigmoid ->
  //       softplus window * edge_weight, clip [0, 5]
  const float w2v = W2[m];
  const float b2v = b2[0];
  float gate[8];
#pragma unroll
  for (int r = 0; r < 8; ++r) {
    float hv = acc[r];
    hv = hv > 0.0f ? hv : 0.1f * hv;       // leaky_relu(0.1)
    float p = hv * w2v;
    p += __shfl_xor(p, 1, 32);
    p += __shfl_xor(p, 2, 32);
    p += __shfl_xor(p, 4, 32);
    p += __shfl_xor(p, 8, 32);             // row-sum within each 16-lane half
    const int   er  = tile * 16 + r + 8 * h;
    const int   erc = er < E ? er : E - 1;
    const float z   = p + b2v;
    const float sg  = 1.0f / (1.0f + __expf(-z));
    const float t4  = 4.0f * (sg - 0.5f);
    float w = ew[erc] * log1pf(__expf(t4));
    w = fminf(fmaxf(w, 0.0f), 5.0f);
    gate[r] = (er < E) ? w : 0.0f;         // mask tail edges
  }

  // exchange gates through LDS (wave-local; CDNA5 split DS counter wait)
#pragma unroll
  for (int r = 0; r < 8; ++r)
    if (m == r) wlds[wave][8 * h + r] = gate[r];
  asm volatile("s_wait_dscnt 0" ::: "memory");
  const float wm = wlds[wave][m];

  // scatter: out[dst] += msg*w ; deg[dst] += |w|
  float* orow = out + (size_t)dst * DFEAT;
#pragma unroll
  for (int c = 0; c < 4; ++c) {
#pragma unroll
    for (int j = 0; j < 8; ++j) {
      atomicAdd(orow + 32 * c + 8 * h + j,      msg[c * 16 + j]     * wm);
      atomicAdd(orow + 32 * c + 16 + 8 * h + j, msg[c * 16 + 8 + j] * wm);
    }
  }
  if (h == 0) atomicAdd(deg + dst, fabsf(wm));
}

// ---------------------------------------------------------------------------
// Kernel 3: out = gelu(out/(deg+eps) + x_proj), exact erf GELU, float4.
// ---------------------------------------------------------------------------
__global__ __launch_bounds__(256) void finalize_kernel(
    const float4* __restrict__ xp, const float* __restrict__ deg,
    float4* __restrict__ out, int total4)
{
  const int i = blockIdx.x * blockDim.x + threadIdx.x;
  if (i >= total4) return;
  const float dn = deg[i >> 5] + 1e-6f;    // 32 float4 per 128-wide row
  float4 o = out[i];
  const float4 p = xp[i];
  float v;
  v = o.x / dn + p.x; o.x = 0.5f * v * (1.0f + erff(v * 0.70710678118654752f));
  v = o.y / dn + p.y; o.y = 0.5f * v * (1.0f + erff(v * 0.70710678118654752f));
  v = o.z / dn + p.z; o.z = 0.5f * v * (1.0f + erff(v * 0.70710678118654752f));
  v = o.w / dn + p.w; o.w = 0.5f * v * (1.0f + erff(v * 0.70710678118654752f));
  out[i] = o;
}

// ---------------------------------------------------------------------------
extern "C" void kernel_launch(void* const* d_in, const int* in_sizes, int n_in,
                              void* d_out, int out_size, void* d_ws, size_t ws_size,
                              hipStream_t stream) {
  const float*     x    = (const float*)d_in[0];
  const float*     st   = (const float*)d_in[1];
  const long long* ei   = (const long long*)d_in[2];   // int64 [2, E]
  const float*     ew   = (const float*)d_in[3];
  const float*     W_in = (const float*)d_in[4];
  const float*     b_in = (const float*)d_in[5];
  const float*     W1   = (const float*)d_in[6];
  const float*     b1   = (const float*)d_in[7];
  const float*     W2   = (const float*)d_in[8];
  const float*     b2   = (const float*)d_in[9];
  (void)n_in; (void)out_size; (void)ws_size;

  const int N = in_sizes[0] / DFEAT;
  const int E = in_sizes[3];

  float* xp  = (float*)d_ws;                 // [N,128] x_proj
  float* deg = xp + (size_t)N * DFEAT;       // [N] degree
  float* out = (float*)d_out;                // accumulator, then final output

  (void)hipMemsetAsync(out, 0, (size_t)N * DFEAT * sizeof(float), stream);
  (void)hipMemsetAsync(deg, 0, (size_t)N * sizeof(float), stream);

  const int mtiles = (N + 15) / 16;
  proj_wmma_kernel<<<mtiles, 256, 0, stream>>>(x, st, W_in, b_in, xp, N);

  const int etiles = (E + 15) / 16;
  edge_wmma_kernel<<<(etiles + 7) / 8, 256, 0, stream>>>(xp, ei, ew, W1, b1, W2, b2,
                                                         out, deg, E, etiles);

  const int total4 = N * (DFEAT / 4);
  finalize_kernel<<<(total4 + 255) / 256, 256, 0, stream>>>((const float4*)xp, deg,
                                                            (float4*)out, total4);
}
